// NerfDecoder_48172353192664
// MI455X (gfx1250) — matrix operand order repaired and checked by reading, not compiled, not run on hardware
//
#include <hip/hip_runtime.h>
#include <hip/hip_bf16.h>

typedef __attribute__((ext_vector_type(16))) _Float16 v16h;
typedef __attribute__((ext_vector_type(8)))  _Float16 v8h;
typedef __attribute__((ext_vector_type(8)))  float    v8f;

#define D_FEAT 90
#define DPAD   96
#define NHEADS 12
#define DH     64
#define INNER  768
#define MLPD   1536
#define KVTOK  128
#define NC     64
#define NF     32
#define MIN_DIST 0.035f
#define MAX_DIST 35.0f
#define PI_F 3.14159265358979323846f

// ---------------------------------------------------------------- WMMA helpers

__device__ __forceinline__ v8f wmma32(v16h a, v16h b, v8f c) {
  return __builtin_amdgcn_wmma_f32_16x16x32_f16(false, a, false, b, (short)0, c,
                                                false, false);
}

// A fragment of row-major MxK matrix (tile 16x32 at row0,k0), per ISA 7.12.2:
// lanes 0-15 hold row M=lane, K = {k0..k0+7, k0+16..k0+23};
// lanes 16-31 hold row M=lane-16, K = {k0+8..k0+15, k0+24..k0+31}.
// Requires 16-byte alignment of (row stride * 2) and k0%8==0 elements.
__device__ __forceinline__ v16h load_a_frag(const _Float16* A, int lda,
                                            int row0, int k0, int lane) {
  int lh = lane & 15, hi = lane >> 4;
  const _Float16* r = A + (size_t)(row0 + lh) * lda + k0 + hi * 8;
  v8h lo = *(const v8h*)r;
  v8h hi8 = *(const v8h*)(r + 16);
  return __builtin_shufflevector(lo, hi8, 0, 1, 2, 3, 4, 5, 6, 7, 8, 9, 10, 11,
                                 12, 13, 14, 15);
}

// B fragment where memory holds B transposed (Bt[n][kk], K contiguous):
// lanes 0-15 hold col N=col0+lane, K={k0..k0+7, k0+16..k0+23}; lanes 16-31 hold K={k0+8..k0+15, k0+24..k0+31}.
__device__ __forceinline__ v16h load_bT_frag(const _Float16* Bt, int ldt,
                                             int k0, int col0, int lane) {
  int lh = lane & 15, hi = lane >> 4;
  const _Float16* p = Bt + (size_t)(col0 + lh) * ldt + k0 + hi * 8;
  v8h lo = *(const v8h*)p;
  v8h hi8 = *(const v8h*)(p + 16);
  return __builtin_shufflevector(lo, hi8, 0, 1, 2, 3, 4, 5, 6, 7, 8, 9, 10, 11,
                                 12, 13, 14, 15);
}

// ---------------------------------------------------------------- prep: f32 -> padded, TRANSPOSED f16 weights
// All f16 weights are stored N-major / K-contiguous so every WMMA B-fragment
// is two contiguous 16-byte loads.

__global__ __launch_bounds__(256) void prep_kernel(
    const float* __restrict__ z, const float* __restrict__ wq,
    const float* __restrict__ wkv, const float* __restrict__ wo,
    const float* __restrict__ w1, const float* __restrict__ w2,
    const float* __restrict__ wc1, const float* __restrict__ wd1,
    _Float16* z16, _Float16* wq16t, _Float16* wkv16t, _Float16* wo16t,
    _Float16* w116t, _Float16* w216t, _Float16* wc116t, _Float16* wd116t) {
  int stride = gridDim.x * blockDim.x;
  int tid0 = blockIdx.x * blockDim.x + threadIdx.x;

  for (int i = tid0; i < 2 * INNER * DPAD; i += stride) {   // wq^T [2][768][96]
    int l = i / (INNER * DPAD), n = (i / DPAD) % INNER, k = i % DPAD;
    wq16t[i] = (_Float16)(k < D_FEAT ? wq[((size_t)l * D_FEAT + k) * INNER + n] : 0.f);
  }
  for (int i = tid0; i < 2 * 2 * INNER * INNER; i += stride) { // wkv^T [2][1536][768]
    int l = i / (2 * INNER * INNER), n = (i / INNER) % (2 * INNER), k = i % INNER;
    wkv16t[i] = (_Float16)wkv[((size_t)l * INNER + k) * (2 * INNER) + n];
  }
  for (int i = tid0; i < 2 * DPAD * INNER; i += stride) {   // wo^T [2][96][768]
    int l = i / (DPAD * INNER), n = (i / INNER) % DPAD, k = i % INNER;
    wo16t[i] = (_Float16)(n < D_FEAT ? wo[((size_t)l * INNER + k) * D_FEAT + n] : 0.f);
  }
  for (int i = tid0; i < 2 * MLPD * DPAD; i += stride) {    // w1^T [2][1536][96]
    int l = i / (MLPD * DPAD), n = (i / DPAD) % MLPD, k = i % DPAD;
    w116t[i] = (_Float16)(k < D_FEAT ? w1[((size_t)l * D_FEAT + k) * MLPD + n] : 0.f);
  }
  for (int i = tid0; i < 2 * DPAD * MLPD; i += stride) {    // w2^T [2][96][1536]
    int l = i / (DPAD * MLPD), n = (i / MLPD) % DPAD, k = i % MLPD;
    w216t[i] = (_Float16)(n < D_FEAT ? w2[((size_t)l * MLPD + k) * D_FEAT + n] : 0.f);
  }
  for (int i = tid0; i < 128 * DPAD; i += stride) {         // wd1^T / wc1^T [128][96]
    int n = i / DPAD, k = i % DPAD;
    wd116t[i] = (_Float16)(k < D_FEAT ? wd1[k * 128 + n] : 0.f);
    wc116t[i] = (_Float16)(k < D_FEAT ? wc1[k * 128 + n] : 0.f);
  }
  for (int i = tid0; i < KVTOK * INNER; i += stride)        // z [128][768]
    z16[i] = (_Float16)z[i];
}

// ---------------------------------------------------------------- kv = z @ wkv  (per layer)
// k stored [head][token][64] (K^T for the score GEMM);
// v stored [head][64][token] (V^T: K-contiguous B for the P.V GEMM).

__global__ __launch_bounds__(128) void kv_kernel(
    const _Float16* __restrict__ z16, const _Float16* __restrict__ wkvlt,
    _Float16* __restrict__ kb, _Float16* __restrict__ vb) {
  int wave = threadIdx.x >> 5, lane = threadIdx.x & 31;
  int t = blockIdx.x * 4 + wave;   // 768 tiles: 8 M-tiles x 96 N-tiles
  int mt = t & 7, nt = t >> 3;
  v8f acc = {};
  for (int ks = 0; ks < INNER; ks += 32) {
    v16h a = load_a_frag(z16, INNER, mt * 16, ks, lane);
    v16h b = load_bT_frag(wkvlt, INNER, ks, nt * 16, lane);
    acc = wmma32(a, b, acc);
  }
  int lh = lane & 15, hi = lane >> 4;
#pragma unroll
  for (int j = 0; j < 8; ++j) {
    int tok = mt * 16 + hi * 8 + j;
    int c = nt * 16 + lh;
    int head = c >> 7, r = c & 127;
    _Float16 val = (_Float16)acc[j];
    if (r < DH) kb[((size_t)head * KVTOK + tok) * DH + r] = val;
    else        vb[((size_t)head * DH + (r - DH)) * KVTOK + tok] = val;
  }
}

// ---------------------------------------------------------------- coords + positional encoding

__global__ __launch_bounds__(256) void posenc_kernel(
    const float* __restrict__ x, const float* __restrict__ rays,
    const float* __restrict__ depths, int spr, float* __restrict__ h, int N) {
  int idx = blockIdx.x * blockDim.x + threadIdx.x;
  if (idx >= N) return;
  int ray = idx / spr;
  float d = depths[idx];
  float* out = h + (size_t)idx * D_FEAT;
#pragma unroll
  for (int j = 0; j < 3; ++j) {
    float c = x[ray * 3 + j] + rays[ray * 3 + j] * d;
    float mult = PI_F;
    for (int o = 0; o < 15; ++o) {
      float sc = c * mult;
      out[15 * j + o] = __sinf(sc);
      out[45 + 15 * j + o] = __cosf(sc);
      mult *= 2.f;
    }
  }
}

__global__ __launch_bounds__(256) void fill_coarse_kernel(float* dc, int total) {
  int i = blockIdx.x * blockDim.x + threadIdx.x;
  if (i >= total) return;
  int s = i % NC;
  float step = (MAX_DIST - MIN_DIST) / NC;
  dc[i] = MIN_DIST + step * s + 0.5f * step;
}

// ---------------------------------------------------------------- fused attention layer
// block = 128 threads (4 waves), one 16-row tile. h += (softmax(QK^T/8) V) Wo + bo.

__global__ __launch_bounds__(128) void attn_kernel(
    float* __restrict__ h, const _Float16* __restrict__ wq16t,
    const _Float16* __restrict__ kb, const _Float16* __restrict__ vb,
    const _Float16* __restrict__ wo16t, const float* __restrict__ lnw,
    const float* __restrict__ lnb, const float* __restrict__ bo) {
  __shared__ _Float16 hn[16 * DPAD];
  __shared__ _Float16 qs[4][16 * DH];
  __shared__ _Float16 ps[4][16 * KVTOK];
  __shared__ _Float16 os[16 * INNER];

  int tid = threadIdx.x, wave = tid >> 5, lane = tid & 31;
  int row0 = blockIdx.x * 16;

  // LayerNorm (90 features) -> hn padded to 96, f16
  if (tid < 16) {
    const float* hr = h + (size_t)(row0 + tid) * D_FEAT;
    float m = 0.f;
    for (int j = 0; j < D_FEAT; ++j) m += hr[j];
    m /= (float)D_FEAT;
    float v = 0.f;
    for (int j = 0; j < D_FEAT; ++j) { float d = hr[j] - m; v += d * d; }
    v /= (float)D_FEAT;
    float inv = rsqrtf(v + 1e-5f);
    for (int j = 0; j < D_FEAT; ++j)
      hn[tid * DPAD + j] = (_Float16)((hr[j] - m) * inv * lnw[j] + lnb[j]);
    for (int j = D_FEAT; j < DPAD; ++j) hn[tid * DPAD + j] = (_Float16)0.f;
  }
  __syncthreads();

  _Float16* qw = qs[wave];
  _Float16* pw = ps[wave];
  int lh = lane & 15, hi = lane >> 4;

  for (int hd = wave; hd < NHEADS; hd += 4) {   // 3 iterations for every wave
    const _Float16* kh = kb + (size_t)hd * KVTOK * DH;   // [token][64]
    const _Float16* vht = vb + (size_t)hd * DH * KVTOK;  // [64][token]

    // q = hn @ wq[:, hd*64 .. +64]   (16 x 64)
#pragma unroll
    for (int nt = 0; nt < 4; ++nt) {
      v8f acc = {};
#pragma unroll
      for (int ks = 0; ks < DPAD; ks += 32) {
        v16h a = load_a_frag(hn, DPAD, 0, ks, lane);
        v16h b = load_bT_frag(wq16t, DPAD, ks, hd * DH + nt * 16, lane);
        acc = wmma32(a, b, acc);
      }
#pragma unroll
      for (int j = 0; j < 8; ++j)
        qw[(hi * 8 + j) * DH + nt * 16 + lh] = (_Float16)acc[j];
    }
    __syncthreads();

    // scores = q @ k^T * (1/sqrt(64))  (16 x 128)
#pragma unroll
    for (int nt = 0; nt < 8; ++nt) {
      v8f acc = {};
#pragma unroll
      for (int ks = 0; ks < DH; ks += 32) {
        v16h a = load_a_frag(qw, DH, 0, ks, lane);
        v16h b = load_bT_frag(kh, DH, ks, nt * 16, lane);
        acc = wmma32(a, b, acc);
      }
#pragma unroll
      for (int j = 0; j < 8; ++j)
        pw[(hi * 8 + j) * KVTOK + nt * 16 + lh] = (_Float16)(acc[j] * 0.125f);
    }
    __syncthreads();

    // softmax over 128 keys; lane r<16 and r+16 split halves of a row
    {
      _Float16* row = pw + lh * KVTOK + hi * 64;
      float mx = -1e30f;
      for (int i = 0; i < 64; ++i) mx = fmaxf(mx, (float)row[i]);
      mx = fmaxf(mx, __shfl_xor(mx, 16));
      float s = 0.f;
      for (int i = 0; i < 64; ++i) {
        float e = __expf((float)row[i] - mx);
        s += e;
        row[i] = (_Float16)e;
      }
      s += __shfl_xor(s, 16);
      float inv = 1.f / s;
      for (int i = 0; i < 64; ++i) row[i] = (_Float16)((float)row[i] * inv);
    }
    __syncthreads();

    // o_head = P @ V   (16 x 64) -> os slice; B = V^T (K-contiguous)
#pragma unroll
    for (int nt = 0; nt < 4; ++nt) {
      v8f acc = {};
#pragma unroll
      for (int ks = 0; ks < KVTOK; ks += 32) {
        v16h a = load_a_frag(pw, KVTOK, 0, ks, lane);
        v16h b = load_bT_frag(vht, KVTOK, ks, nt * 16, lane);
        acc = wmma32(a, b, acc);
      }
#pragma unroll
      for (int j = 0; j < 8; ++j)
        os[(hi * 8 + j) * INNER + hd * DH + nt * 16 + lh] = (_Float16)acc[j];
    }
    __syncthreads();
  }

  // h += o @ wo + bo   (K = 768, 6 col tiles of padded 96)
  for (int ct = wave; ct < 6; ct += 4) {
    v8f acc = {};
    for (int ks = 0; ks < INNER; ks += 32) {
      v16h a = load_a_frag(os, INNER, 0, ks, lane);
      v16h b = load_bT_frag(wo16t, INNER, ks, ct * 16, lane);
      acc = wmma32(a, b, acc);
    }
    int col = ct * 16 + lh;
    if (col < D_FEAT) {
#pragma unroll
      for (int j = 0; j < 8; ++j) {
        size_t r = (size_t)(row0 + hi * 8 + j);
        h[r * D_FEAT + col] += acc[j] + bo[col];
      }
    }
  }
}

// ---------------------------------------------------------------- fused MLP layer
// block = 1 wave, one 16-row tile. h += gelu(LN(h)@w1+b1) @ w2 + b2,
// streaming the 1536-wide hidden in 32-column chunks through LDS.

__global__ __launch_bounds__(32) void mlp_kernel(
    float* __restrict__ h, const _Float16* __restrict__ w116t,
    const float* __restrict__ b1, const _Float16* __restrict__ w216t,
    const float* __restrict__ b2, const float* __restrict__ lnw,
    const float* __restrict__ lnb) {
  __shared__ _Float16 hn[16 * DPAD];
  __shared__ _Float16 us[16 * 32];

  int lane = threadIdx.x;
  int lh = lane & 15, hi = lane >> 4;
  int row0 = blockIdx.x * 16;

  // cooperative LayerNorm: (row = lane&15, half = lane>>4) covers 45 cols each
  {
    const float* hr = h + (size_t)(row0 + lh) * D_FEAT;
    int c0 = hi * 45;
    float s = 0.f, s2 = 0.f;
    for (int j = c0; j < c0 + 45; ++j) { float x = hr[j]; s += x; s2 += x * x; }
    s += __shfl_xor(s, 16);
    s2 += __shfl_xor(s2, 16);
    float m = s / (float)D_FEAT;
    float var = s2 / (float)D_FEAT - m * m;
    float inv = rsqrtf(var + 1e-5f);
    for (int j = c0; j < c0 + 45; ++j)
      hn[lh * DPAD + j] = (_Float16)((hr[j] - m) * inv * lnw[j] + lnb[j]);
    if (hi)
      for (int j = D_FEAT; j < DPAD; ++j) hn[lh * DPAD + j] = (_Float16)0.f;
  }
  __syncthreads();

  v8f acc[6] = {};
  for (int ch = 0; ch < MLPD; ch += 32) {
    // u = gelu(hn @ w1[:, ch..ch+32] + b1)
#pragma unroll
    for (int t = 0; t < 2; ++t) {
      v8f u = {};
#pragma unroll
      for (int ks = 0; ks < DPAD; ks += 32) {
        v16h a = load_a_frag(hn, DPAD, 0, ks, lane);
        v16h b = load_bT_frag(w116t, DPAD, ks, ch + t * 16, lane);
        u = wmma32(a, b, u);
      }
#pragma unroll
      for (int j = 0; j < 8; ++j) {
        float xv = u[j] + b1[ch + t * 16 + lh];
        float g = 0.5f * xv * (1.0f + erff(xv * 0.70710678118654752f)); // exact gelu
        us[(hi * 8 + j) * 32 + t * 16 + lh] = (_Float16)g;
      }
    }
    __syncthreads();
    // acc += u (16x32) @ w2[ch..ch+32, 0..96]
    v16h a = load_a_frag(us, 32, 0, 0, lane);
#pragma unroll
    for (int nt = 0; nt < 6; ++nt) {
      v16h b = load_bT_frag(w216t, MLPD, ch, nt * 16, lane);
      acc[nt] = wmma32(a, b, acc[nt]);
    }
    __syncthreads();
  }

#pragma unroll
  for (int nt = 0; nt < 6; ++nt) {
    int col = nt * 16 + lh;
    if (col < D_FEAT) {
#pragma unroll
      for (int j = 0; j < 8; ++j) {
        size_t r = (size_t)(row0 + hi * 8 + j);
        h[r * D_FEAT + col] += acc[nt][j] + b2[col];
      }
    }
  }
}

// ---------------------------------------------------------------- density & color heads

__global__ __launch_bounds__(32) void heads_kernel(
    const float* __restrict__ h, const _Float16* __restrict__ wd116t,
    const float* __restrict__ bd1, const float* __restrict__ wd2,
    const float* __restrict__ bd2, const _Float16* __restrict__ wc116t,
    const float* __restrict__ bc1, const float* __restrict__ wc2,
    const float* __restrict__ bc2, float* __restrict__ dens,
    float* __restrict__ col) {
  __shared__ _Float16 hs[16 * DPAD];
  __shared__ float as_[16 * 128];
  __shared__ float cs[16 * 128];

  int lane = threadIdx.x;
  int lh = lane & 15, hi = lane >> 4;
  int row0 = blockIdx.x * 16;

  {
    const float* hr = h + (size_t)(row0 + lh) * D_FEAT;
    int c0 = hi * 45;
    for (int j = c0; j < c0 + 45; ++j) hs[lh * DPAD + j] = (_Float16)hr[j];
    if (hi)
      for (int j = D_FEAT; j < DPAD; ++j) hs[lh * DPAD + j] = (_Float16)0.f;
  }
  __syncthreads();

#pragma unroll
  for (int nt = 0; nt < 8; ++nt) {
    v8f ad = {}, ac = {};
#pragma unroll
    for (int ks = 0; ks < DPAD; ks += 32) {
      v16h a = load_a_frag(hs, DPAD, 0, ks, lane);
      v16h bd = load_bT_frag(wd116t, DPAD, ks, nt * 16, lane);
      v16h bc = load_bT_frag(wc116t, DPAD, ks, nt * 16, lane);
      ad = wmma32(a, bd, ad);
      ac = wmma32(a, bc, ac);
    }
#pragma unroll
    for (int j = 0; j < 8; ++j) {
      int cc = nt * 16 + lh;
      as_[(hi * 8 + j) * 128 + cc] = fmaxf(0.f, ad[j] + bd1[cc]);
      cs[(hi * 8 + j) * 128 + cc] = fmaxf(0.f, ac[j] + bc1[cc]);
    }
  }
  __syncthreads();

  if (lane < 16) {
    float x = bd2[0];
    for (int k = 0; k < 128; ++k) x += as_[lane * 128 + k] * wd2[k];
    dens[row0 + lane] = (x > 20.f) ? x : log1pf(__expf(x));  // softplus
  } else {
    int r = lane - 16;
    for (int j = 0; j < 3; ++j) {
      float y = bc2[j];
      for (int k = 0; k < 128; ++k) y += cs[r * 128 + k] * wc2[k * 3 + j];
      col[(size_t)(row0 + r) * 3 + j] = 1.f / (1.f + __expf(-y));  // sigmoid
    }
  }
}

// ---------------------------------------------------------------- volume rendering

__global__ __launch_bounds__(64) void render_kernel(
    const float* __restrict__ dens, const float* __restrict__ colbuf,
    const float* __restrict__ depths, int spr, float* __restrict__ out_rgb,
    float* __restrict__ out_depth, float* __restrict__ wts, int R) {
  int ray = blockIdx.x * blockDim.x + threadIdx.x;
  if (ray >= R) return;
  const float* dn = dens + (size_t)ray * spr;
  const float* cl = colbuf + (size_t)ray * spr * 3;
  const float* dp = depths + (size_t)ray * spr;
  float trans = 1.f, acc = 0.f, r = 0.f, g = 0.f, b = 0.f, dep = 0.f;
  for (int s = 0; s < spr; ++s) {
    float seg = (s + 1 < spr) ? (dp[s + 1] - dp[s]) : 1e10f;
    float alpha = 1.f - __expf(-dn[s] * seg);
    float w = alpha * trans;
    trans *= (1.f - alpha + 1e-10f);
    acc += w;
    r += w * cl[s * 3 + 0];
    g += w * cl[s * 3 + 1];
    b += w * cl[s * 3 + 2];
    dep += w * dp[s];
    if (wts) wts[(size_t)ray * spr + s] = w;
  }
  float inv = 1.f / (acc + 1e-8f);
  // comp = 1 - acc + rgb * acc
  out_rgb[ray * 3 + 0] = 1.f - acc + (r * inv) * acc;
  out_rgb[ray * 3 + 1] = 1.f - acc + (g * inv) * acc;
  out_rgb[ray * 3 + 2] = 1.f - acc + (b * inv) * acc;
  out_depth[ray] = dep * inv;
}

// ---------------------------------------------------------------- inverse-CDF fine sampling

__global__ __launch_bounds__(64) void samplepdf_kernel(
    const float* __restrict__ wts, float* __restrict__ fine, int R) {
  int ray = blockIdx.x * blockDim.x + threadIdx.x;
  if (ray >= R) return;
  float step = (MAX_DIST - MIN_DIST) / NC;
  float cdf[NC];     // 63 used
  float w[NC];       // 62 used
  float wsum = 0.f;
  for (int i = 0; i < NC - 2; ++i) {
    w[i] = wts[(size_t)ray * NC + 1 + i] + 1e-5f;
    wsum += w[i];
  }
  float inv = 1.f / wsum;
  cdf[0] = 0.f;
  for (int i = 0; i < NC - 2; ++i) cdf[i + 1] = cdf[i] + w[i] * inv;
  const int nb = NC - 1;  // 63 bins/cdf entries; bins[i] = MIN + step*(i+1)
  for (int k = 0; k < NF; ++k) {
    float u = (k + 0.5f) / (float)NF;
    int lo = 0, hih = nb;
    while (lo < hih) {
      int mid = (lo + hih) >> 1;
      if (cdf[mid] <= u) lo = mid + 1; else hih = mid;
    }
    int below = min(max(lo - 1, 0), nb - 1);
    int above = min(lo, nb - 1);
    float cb = cdf[below], ca = cdf[above];
    float bb = MIN_DIST + step * (below + 1);
    float ba = MIN_DIST + step * (above + 1);
    float denom = (ca - cb < 1e-5f) ? 1.f : (ca - cb);
    fine[(size_t)ray * NF + k] = bb + (u - cb) / denom * (ba - bb);
  }
}

// merge sorted coarse (analytic) with sorted fine samples -> depths_agg [R][96]
__global__ __launch_bounds__(64) void merge_kernel(
    const float* __restrict__ fine, float* __restrict__ dagg, int R) {
  int ray = blockIdx.x * blockDim.x + threadIdx.x;
  if (ray >= R) return;
  const float* f = fine + (size_t)ray * NF;
  float* o = dagg + (size_t)ray * (NC + NF);
  float step = (MAX_DIST - MIN_DIST) / NC;
  int i = 0, j = 0;
  for (int k = 0; k < NC + NF; ++k) {
    float cv = (i < NC) ? (MIN_DIST + step * i + 0.5f * step) : 3.4e38f;
    float fv = (j < NF) ? f[j] : 3.4e38f;
    if (cv <= fv) { o[k] = cv; ++i; } else { o[k] = fv; ++j; }
  }
}

// ---------------------------------------------------------------- host

extern "C" void kernel_launch(void* const* d_in, const int* in_sizes, int n_in,
                              void* d_out, int out_size, void* d_ws,
                              size_t ws_size, hipStream_t stream) {
  const float* z    = (const float*)d_in[0];
  const float* x    = (const float*)d_in[1];
  const float* rays = (const float*)d_in[2];
  const float* ln1w = (const float*)d_in[3];
  const float* ln1b = (const float*)d_in[4];
  const float* wq   = (const float*)d_in[5];
  const float* wkv  = (const float*)d_in[6];
  const float* wo   = (const float*)d_in[7];
  const float* bo   = (const float*)d_in[8];
  const float* ln2w = (const float*)d_in[9];
  const float* ln2b = (const float*)d_in[10];
  const float* w1   = (const float*)d_in[11];
  const float* b1   = (const float*)d_in[12];
  const float* w2   = (const float*)d_in[13];
  const float* b2   = (const float*)d_in[14];
  const float* wc1  = (const float*)d_in[15];
  const float* bc1  = (const float*)d_in[16];
  const float* wc2  = (const float*)d_in[17];
  const float* bc2  = (const float*)d_in[18];
  const float* wd1  = (const float*)d_in[19];
  const float* bd1  = (const float*)d_in[20];
  const float* wd2  = (const float*)d_in[21];
  const float* bd2  = (const float*)d_in[22];

  const int R = in_sizes[1] / 3;  // B==1
  const int N1 = R * NC;
  const int N2 = R * (NC + NF);

  char* ws = (char*)d_ws;
  size_t off = 0;
  auto alloc = [&](size_t bytes) -> void* {
    void* p = ws + off;
    off = (off + bytes + 255) & ~(size_t)255;
    return p;
  };

  _Float16* wq16t  = (_Float16*)alloc((size_t)2 * INNER * DPAD * 2);
  _Float16* wkv16t = (_Float16*)alloc((size_t)2 * 2 * INNER * INNER * 2);
  _Float16* wo16t  = (_Float16*)alloc((size_t)2 * DPAD * INNER * 2);
  _Float16* w116t  = (_Float16*)alloc((size_t)2 * MLPD * DPAD * 2);
  _Float16* w216t  = (_Float16*)alloc((size_t)2 * DPAD * MLPD * 2);
  _Float16* wd116t = (_Float16*)alloc((size_t)128 * DPAD * 2);
  _Float16* wc116t = (_Float16*)alloc((size_t)128 * DPAD * 2);
  _Float16* z16    = (_Float16*)alloc((size_t)KVTOK * INNER * 2);
  _Float16* kb     = (_Float16*)alloc((size_t)2 * NHEADS * KVTOK * DH * 2);
  _Float16* vb     = (_Float16*)alloc((size_t)2 * NHEADS * DH * KVTOK * 2);
  float* hbuf      = (float*)alloc((size_t)N2 * D_FEAT * 4);
  float* densb     = (float*)alloc((size_t)N2 * 4);
  float* colb      = (float*)alloc((size_t)N2 * 3 * 4);
  float* dcoarse   = (float*)alloc((size_t)R * NC * 4);
  float* wtsb      = (float*)alloc((size_t)R * NC * 4);
  float* fineb     = (float*)alloc((size_t)R * NF * 4);
  float* dagg      = (float*)alloc((size_t)R * (NC + NF) * 4);
  (void)ws_size; (void)n_in; (void)out_size;

  float* out = (float*)d_out;
  float* fine_rgb   = out;
  float* coarse_rgb = out + (size_t)3 * R;
  float* fine_dep   = out + (size_t)6 * R;
  float* coarse_dep = out + (size_t)7 * R;

  prep_kernel<<<2048, 256, 0, stream>>>(z, wq, wkv, wo, w1, w2, wc1, wd1, z16,
                                        wq16t, wkv16t, wo16t, w116t, w216t,
                                        wc116t, wd116t);
  for (int l = 0; l < 2; ++l)
    kv_kernel<<<192, 128, 0, stream>>>(
        z16, wkv16t + (size_t)l * 2 * INNER * INNER,
        kb + (size_t)l * NHEADS * KVTOK * DH,
        vb + (size_t)l * NHEADS * DH * KVTOK);

  fill_coarse_kernel<<<(R * NC + 255) / 256, 256, 0, stream>>>(dcoarse, R * NC);

  for (int pass = 0; pass < 2; ++pass) {
    int N = pass ? N2 : N1;
    int spr = pass ? (NC + NF) : NC;
    const float* dep = pass ? dagg : dcoarse;

    posenc_kernel<<<(N + 255) / 256, 256, 0, stream>>>(x, rays, dep, spr, hbuf, N);

    for (int l = 0; l < 2; ++l) {
      attn_kernel<<<N / 16, 128, 0, stream>>>(
          hbuf, wq16t + (size_t)l * INNER * DPAD,
          kb + (size_t)l * NHEADS * KVTOK * DH,
          vb + (size_t)l * NHEADS * DH * KVTOK,
          wo16t + (size_t)l * DPAD * INNER, ln1w + l * D_FEAT,
          ln1b + l * D_FEAT, bo + l * D_FEAT);
      mlp_kernel<<<N / 16, 32, 0, stream>>>(
          hbuf, w116t + (size_t)l * MLPD * DPAD, b1 + l * MLPD,
          w216t + (size_t)l * DPAD * MLPD, b2 + l * D_FEAT, ln2w + l * D_FEAT,
          ln2b + l * D_FEAT);
    }
    heads_kernel<<<N / 16, 32, 0, stream>>>(hbuf, wd116t, bd1, wd2, bd2,
                                            wc116t, bc1, wc2, bc2, densb,
                                            colb);
    if (pass == 0) {
      render_kernel<<<(R + 63) / 64, 64, 0, stream>>>(
          densb, colb, dcoarse, NC, coarse_rgb, coarse_dep, wtsb, R);
      samplepdf_kernel<<<(R + 63) / 64, 64, 0, stream>>>(wtsb, fineb, R);
      merge_kernel<<<(R + 63) / 64, 64, 0, stream>>>(fineb, dagg, R);
    } else {
      render_kernel<<<(R + 63) / 64, 64, 0, stream>>>(
          densb, colb, dagg, NC + NF, fine_rgb, fine_dep, nullptr, R);
    }
  }
}
